// RNN_37632503447944
// MI455X (gfx1250) — matrix operand order repaired and checked by reading, not compiled, not run on hardware
//
#include <hip/hip_runtime.h>
#include <hip/hip_bf16.h>
#include <math.h>

#define Bsz 16
#define Ssz 256
#define Esz 512
#define Hsz 1024
#define Vsz 32000
#define MR  (Ssz * Bsz)   // 4096 rows, row index sb = s*16 + b

typedef __attribute__((ext_vector_type(16))) __bf16 v16bf;
typedef __attribute__((ext_vector_type(8)))  float  v8f;
typedef __attribute__((ext_vector_type(4)))  unsigned int u32x4;
typedef __attribute__((ext_vector_type(8)))  int i32x8;
typedef __attribute__((ext_vector_type(4)))  int i32x4;

union Frag { u32x4 u[2]; v16bf v; };

__device__ __forceinline__ unsigned short f2bf(float f) {
  unsigned u = __float_as_uint(f);
  u += 0x7FFFu + ((u >> 16) & 1u);   // round-to-nearest-even
  return (unsigned short)(u >> 16);
}

// Padded LDS layout for staged tiles, in 16B (u32x4) units:
// one 1024-elem bf16 row = 128 units + 2 pad units (16B pad per 1024B stored,
// matching TDM pad_interval=256dw / pad_amount=4dw) -> row stride 130 units,
// second k-half (k>=512) starts at +65 units.
#define ROW_U   130
#define HALF_U  65

// Issue a TDM load of `rows` x 1024 bf16 (contiguous in global) into LDS
// offset 0 with the padded layout above. rows*1024 elems split as a 2D tile
// of 16 x (rows*64) elements so tile_dim0 fits its 16-bit field.
__device__ __forceinline__ void tdm_stage_rows(const unsigned short* gbase, int rows) {
#if __has_builtin(__builtin_amdgcn_tensor_load_to_lds) && __has_builtin(__builtin_amdgcn_s_wait_tensorcnt)
  unsigned long long ga = (unsigned long long)(const void*)gbase;
  unsigned dim0 = (unsigned)(rows * 64);    // elements per tile row (<=8192)
  u32x4 g0;
  g0[0] = 1u;                                // count=1 (valid user descriptor)
  g0[1] = 0u;                                // lds_addr = 0 (single LDS object)
  g0[2] = (unsigned)(ga & 0xFFFFFFFFull);
  g0[3] = (unsigned)((ga >> 32) & 0x01FFFFFFull) | (2u << 30);  // type=2
  i32x8 g1;
  g1[0] = (int)((1u << 16) | (1u << 20) | (7u << 22) | (3u << 25)); // 2B, pad 16B/1024B
  g1[1] = (int)(dim0 << 16);     // tensor_dim0[15:0]
  g1[2] = (int)(16u << 16);      // tensor_dim0 hi=0 | tensor_dim1=16
  g1[3] = (int)(dim0 << 16);     // tensor_dim1 hi=0 | tile_dim0=dim0
  g1[4] = 16;                    // tile_dim1=16, tile_dim2=0
  g1[5] = (int)dim0;             // tensor_dim0_stride
  g1[6] = 0;
  g1[7] = 0;
  i32x4 g2 = {0, 0, 0, 0};
  i32x4 g3 = {0, 0, 0, 0};
  i32x8 g4 = {0, 0, 0, 0, 0, 0, 0, 0};
  __builtin_amdgcn_tensor_load_to_lds(g0, g1, g2, g3, g4, 0);
  __builtin_amdgcn_s_wait_tensorcnt(0);
#else
  (void)gbase; (void)rows;
#endif
}

__global__ void k_init_bar(int* bar) {
  if (threadIdx.x < 8) bar[threadIdx.x] = 0;
}

__global__ void k_f2bf(const float* __restrict__ src, unsigned short* __restrict__ dst, int n) {
  int i = blockIdx.x * blockDim.x + threadIdx.x;
  int stride = gridDim.x * blockDim.x;
  for (; i < n; i += stride) dst[i] = f2bf(src[i]);
}

// x gathered as bf16, row sb = s*B+b (matches einsum 'bse,he->sbh' row order)
__global__ void k_gather(const int* __restrict__ ids, const float* __restrict__ emb,
                         unsigned short* __restrict__ xg) {
  int i = blockIdx.x * blockDim.x + threadIdx.x;
  int stride = gridDim.x * blockDim.x;
  const int n = MR * Esz;
  for (; i < n; i += stride) {
    int sb = i >> 9;            // / Esz
    int e  = i & (Esz - 1);
    int s = sb >> 4, b = sb & 15;
    int tok = ids[b * Ssz + s]; // x_ids is [B,S]
    xg[i] = f2bf(emb[(size_t)tok * Esz + e]);
  }
}

// out[M,N] = A[M,K](bf16) @ W[N,K]^T(bf16) + bias[N]   (fp32 out)
// one wave per 16x16 tile; K-loop of v_wmma_f32_16x16x32_bf16
__global__ void k_gemm_bias(const unsigned short* __restrict__ A,
                            const unsigned short* __restrict__ W,
                            const float* __restrict__ bias,
                            float* __restrict__ out,
                            int Mtiles, int Ntiles, int K) {
  int wave = (int)((blockIdx.x * blockDim.x + threadIdx.x) >> 5);
  if (wave >= Mtiles * Ntiles) return;
  int lane = threadIdx.x & 31;
  int lrow = lane & 15;
  int lhi  = lane >> 4;
  int mt = wave / Ntiles, nt = wave % Ntiles;
  const unsigned short* arow = A + (size_t)(mt * 16 + lrow) * K;
  const unsigned short* brow = W + (size_t)(nt * 16 + lrow) * K;
  v8f acc = {};
  for (int k0 = 0; k0 < K; k0 += 32) {
    Frag fa, fb;
    int ca = k0 + (lhi << 3);    // A: lane-group splits K by 8, second half +16
    int cb = k0 + (lhi << 3);    // B: lane-group splits K by 8, second half +16
    fa.u[0] = *(const u32x4*)(arow + ca);
    fa.u[1] = *(const u32x4*)(arow + ca + 16);
    fb.u[0] = *(const u32x4*)(brow + cb);
    fb.u[1] = *(const u32x4*)(brow + cb + 16);
    acc = __builtin_amdgcn_wmma_f32_16x16x32_bf16(false, fa.v, false, fb.v,
                                                  (short)0, acc, false, false);
  }
  int N = Ntiles * 16;
  int n = nt * 16 + lrow;
  float bv = bias[n];
  for (int r = 0; r < 8; ++r) {
    int m = mt * 16 + lhi * 8 + r;     // C/D layout: M = r (+8 for hi lanes)
    out[(size_t)m * N + n] = acc[r] + bv;
  }
}

// device-scope grid barrier with monotonic counter (bar zeroed each launch)
__device__ __forceinline__ void grid_barrier(int* bar, int nblocks, int step) {
  __threadfence();
  __syncthreads();
  if (threadIdx.x == 0) {
    __hip_atomic_fetch_add(bar, 1, __ATOMIC_ACQ_REL, __HIP_MEMORY_SCOPE_AGENT);
    int target = nblocks * (step + 1);
    while (__hip_atomic_load(bar, __ATOMIC_ACQUIRE, __HIP_MEMORY_SCOPE_AGENT) < target)
      __builtin_amdgcn_s_sleep(1);
  }
  __syncthreads();
}

// persistent RNN scan: h_s = tanh(xpre_s + h_{s-1} @ Whh^T + b); hs rows sb=s*16+b
// launch <<<8,256>>>: 64 waves, wave -> one 16-wide N tile, M=16 (batch).
// Each block's 128 rows of Whh (256KB bf16) are TDM-staged into LDS once and
// reused across all 256 steps: per-step chain = h_prev loads + 32 WMMAs only.
#define SC_ROWS 128
__global__ void __launch_bounds__(256) k_scan(const float* __restrict__ xpre,
                                              const unsigned short* __restrict__ Whh,
                                              const float* __restrict__ bias,
                                              unsigned short* __restrict__ hs,
                                              int* __restrict__ bar) {
  __shared__ u32x4 shB[SC_ROWS * ROW_U + 1];   // ~266KB of 320KB LDS
  int lane = threadIdx.x & 31;
  int lrow = lane & 15;
  int lhi  = lane >> 4;
  int wloc = threadIdx.x >> 5;                    // 0..7 local wave
  int nt   = blockIdx.x * 8 + wloc;               // 0..63 global N tile
  const unsigned short* Bbase = Whh + (size_t)blockIdx.x * SC_ROWS * Hsz;
#if __has_builtin(__builtin_amdgcn_tensor_load_to_lds) && __has_builtin(__builtin_amdgcn_s_wait_tensorcnt)
  if (wloc == 0) tdm_stage_rows(Bbase, SC_ROWS);
#else
  for (int idx = threadIdx.x; idx < SC_ROWS * 128; idx += blockDim.x) {
    int row = idx >> 7;
    int ku  = idx & 127;                         // 16B unit within row
    shB[row * ROW_U + ku + (ku >= 64 ? 1 : 0)] =
        *(const u32x4*)(Bbase + (size_t)row * Hsz + (ku << 3));
  }
#endif
  __syncthreads();

  int rloc = wloc * 16 + lrow;                    // local B row in LDS
  int rbase = rloc * ROW_U;
  int n = nt * 16 + lrow;
  float bv = bias[n];
  for (int s = 0; s < Ssz; ++s) {
    v8f acc = {};
    if (s > 0) {
      const unsigned short* arow = hs + ((size_t)(s - 1) * 16 + lrow) * Hsz;
      // split at k=512: pad shift is constant per half, no per-iter select
#pragma unroll
      for (int half = 0; half < 2; ++half) {
        const unsigned short* ah = arow + half * 512;
        int hb = rbase + half * HALF_U;
#pragma unroll
        for (int k0 = 0; k0 < 512; k0 += 32) {
          Frag fa, fb;
          int ca = k0 + (lhi << 3);
          int cu = hb + (k0 >> 3) + lhi;  // (k0 + lhi*8)/8 units
          fa.u[0] = *(const u32x4*)(ah + ca);
          fa.u[1] = *(const u32x4*)(ah + ca + 16);
          fb.u[0] = shB[cu];
          fb.u[1] = shB[cu + 2];
          acc = __builtin_amdgcn_wmma_f32_16x16x32_bf16(false, fa.v, false, fb.v,
                                                        (short)0, acc, false, false);
        }
      }
    }
    for (int r = 0; r < 8; ++r) {
      int row = s * 16 + lhi * 8 + r;
      float v = tanhf(acc[r] + xpre[(size_t)row * Hsz + n] + bv);
      hs[(size_t)row * Hsz + n] = f2bf(v);
    }
    grid_barrier(bar, (int)gridDim.x, s);
  }
}

// fc GEMM: A tile (16x1024 bf16, 32KB) TDM-staged into LDS with padding;
// each wave computes 4 16x16 N tiles against L2-resident bf16 W_fc.
__global__ void k_fc(const unsigned short* __restrict__ h2s,
                     const unsigned short* __restrict__ Wfc,
                     const float* __restrict__ bfc,
                     float* __restrict__ out) {
  __shared__ u32x4 shA[16 * ROW_U + 1];
  int mt   = blockIdx.y;             // 0..255 (M tile over sb rows)
  int wave = threadIdx.x >> 5;
  int lane = threadIdx.x & 31;
  int lrow = lane & 15;
  int lhi  = lane >> 4;
  const unsigned short* Abase = h2s + (size_t)mt * 16 * Hsz;
#if __has_builtin(__builtin_amdgcn_tensor_load_to_lds) && __has_builtin(__builtin_amdgcn_s_wait_tensorcnt)
  if (wave == 0) tdm_stage_rows(Abase, 16);
#else
  for (int idx = threadIdx.x; idx < 16 * 128; idx += blockDim.x) {
    int row = idx >> 7;
    int ku  = idx & 127;
    shA[row * ROW_U + ku + (ku >= 64 ? 1 : 0)] =
        *(const u32x4*)(Abase + (size_t)row * Hsz + (ku << 3));
  }
#endif
  __syncthreads();

  int ntb = blockIdx.x * 32 + wave * 4;   // 4 N-tiles per wave, Ntiles=2000
  v8f acc[4] = {{}, {}, {}, {}};
  const unsigned short* brow[4];
  bool valid[4];
  for (int t = 0; t < 4; ++t) {
    valid[t] = (ntb + t) < (Vsz / 16);
    brow[t]  = Wfc + (size_t)((ntb + t) * 16 + lrow) * Hsz;
  }
  int abase = lrow * ROW_U;
  for (int half = 0; half < 2; ++half) {
    int ab = abase + half * HALF_U;
    for (int k0 = 0; k0 < 512; k0 += 32) {
      Frag fa;
      int au = ab + (k0 >> 3) + lhi;         // (k0 + lhi*8)/8 units
      fa.u[0] = shA[au];
      fa.u[1] = shA[au + 2];                 // +16 elements
      int cb = half * 512 + k0 + (lhi << 3);
      for (int t = 0; t < 4; ++t) {
        if (!valid[t]) continue;                 // wave-uniform branch
        Frag fb;
        fb.u[0] = *(const u32x4*)(brow[t] + cb);
        fb.u[1] = *(const u32x4*)(brow[t] + cb + 16);
        if (cb + 64 < Hsz) __builtin_prefetch(brow[t] + cb + 64, 0, 1);
        acc[t] = __builtin_amdgcn_wmma_f32_16x16x32_bf16(false, fa.v, false, fb.v,
                                                         (short)0, acc[t], false, false);
      }
    }
  }
  for (int t = 0; t < 4; ++t) {
    if (!valid[t]) continue;
    int n = (ntb + t) * 16 + lrow;
    float bv = bfc[n];
    for (int r = 0; r < 8; ++r) {
      int m = mt * 16 + lhi * 8 + r;           // sb = s*16 + b
      int orow = (m & 15) * Ssz + (m >> 4);    // output [B,S,V] row = b*S + s
      out[(size_t)orow * Vsz + n] = acc[t][r] + bv;
    }
  }
}

// in-place row softmax over V=32000: online (max,sum) pass + normalize pass
__global__ void k_softmax(float* __restrict__ out) {
  __shared__ float sm[256], sl[256];
  float* base = out + (size_t)blockIdx.x * Vsz;
  int tid = threadIdx.x;
  float m = -3.4e38f, l = 0.f;
  for (int i = tid; i < Vsz; i += 256) {
    float x = base[i];
    float nm = fmaxf(m, x);
    l = l * __expf(m - nm) + __expf(x - nm);
    m = nm;
  }
  sm[tid] = m; sl[tid] = l;
  __syncthreads();
  for (int o = 128; o > 0; o >>= 1) {
    if (tid < o) {
      float m2 = sm[tid + o], l2 = sl[tid + o];
      float M = fmaxf(sm[tid], m2);
      sl[tid] = sl[tid] * __expf(sm[tid] - M) + l2 * __expf(m2 - M);
      sm[tid] = M;
    }
    __syncthreads();
  }
  float M = sm[0];
  float rinv = 1.0f / sl[0];
  for (int i = tid; i < Vsz; i += 256)
    base[i] = __expf(base[i] - M) * rinv;
}

extern "C" void kernel_launch(void* const* d_in, const int* in_sizes, int n_in,
                              void* d_out, int out_size, void* d_ws, size_t ws_size,
                              hipStream_t stream) {
  (void)in_sizes; (void)n_in; (void)out_size; (void)ws_size;
  const int*   x_ids = (const int*)d_in[0];
  const float* emb   = (const float*)d_in[1];
  const float* W_ih1 = (const float*)d_in[2];
  const float* b_ih1 = (const float*)d_in[3];
  const float* W_hh1 = (const float*)d_in[4];
  const float* b_hh1 = (const float*)d_in[5];
  const float* W_ih2 = (const float*)d_in[6];
  const float* b_ih2 = (const float*)d_in[7];
  const float* W_hh2 = (const float*)d_in[8];
  const float* b_hh2 = (const float*)d_in[9];
  const float* W_fc  = (const float*)d_in[10];
  const float* b_fc  = (const float*)d_in[11];
  float* out = (float*)d_out;

  char* ws = (char*)d_ws;
  size_t off = 0;
  auto carve = [&](size_t bytes) -> void* {
    off = (off + 255) & ~(size_t)255;
    void* p = ws + off;
    off += bytes;
    return p;
  };
  unsigned short* xg    = (unsigned short*)carve((size_t)MR  * Esz * 2);
  unsigned short* wih1b = (unsigned short*)carve((size_t)Hsz * Esz * 2);
  unsigned short* whh1b = (unsigned short*)carve((size_t)Hsz * Hsz * 2);
  unsigned short* wih2b = (unsigned short*)carve((size_t)Hsz * Hsz * 2);
  unsigned short* whh2b = (unsigned short*)carve((size_t)Hsz * Hsz * 2);
  unsigned short* wfcb  = (unsigned short*)carve((size_t)Vsz * Hsz * 2);
  float*          x1    = (float*)carve((size_t)MR * Hsz * 4);
  float*          x2    = (float*)carve((size_t)MR * Hsz * 4);
  unsigned short* h1s   = (unsigned short*)carve((size_t)MR * Hsz * 2);
  unsigned short* h2s   = (unsigned short*)carve((size_t)MR * Hsz * 2);
  int*            bar   = (int*)carve(64);

  k_init_bar<<<1, 64, 0, stream>>>(bar);
  k_f2bf<<<512,  256, 0, stream>>>(W_ih1, wih1b, Hsz * Esz);
  k_f2bf<<<1024, 256, 0, stream>>>(W_hh1, whh1b, Hsz * Hsz);
  k_f2bf<<<1024, 256, 0, stream>>>(W_ih2, wih2b, Hsz * Hsz);
  k_f2bf<<<1024, 256, 0, stream>>>(W_hh2, whh2b, Hsz * Hsz);
  k_f2bf<<<8192, 256, 0, stream>>>(W_fc,  wfcb,  Vsz * Hsz);
  k_gather<<<2048, 256, 0, stream>>>(x_ids, emb, xg);
  // x1 = xg @ W_ih1^T + b_ih1 : M=4096 K=512 N=1024
  k_gemm_bias<<<2048, 256, 0, stream>>>(xg, wih1b, b_ih1, x1, MR / 16, Hsz / 16, Esz);
  k_scan<<<8, 256, 0, stream>>>(x1, whh1b, b_hh1, h1s, bar + 0);
  // x2 = h1s @ W_ih2^T + b_ih2 : M=4096 K=1024 N=1024
  k_gemm_bias<<<2048, 256, 0, stream>>>(h1s, wih2b, b_ih2, x2, MR / 16, Hsz / 16, Hsz);
  k_scan<<<8, 256, 0, stream>>>(x2, whh2b, b_hh2, h2s, bar + 1);
  dim3 fcgrid(63, 256);  // 63*32=2016 >= 2000 N-tiles, 256 M-tiles
  k_fc<<<fcgrid, 256, 0, stream>>>(h2s, wfcb, b_fc, out);
  k_softmax<<<4096, 256, 0, stream>>>(out);
}
